// MoELoRALayer_17575006175840
// MI455X (gfx1250) — matrix operand order repaired and checked by reading, not compiled, not run on hardware
//
#include <hip/hip_runtime.h>
#include <hip/hip_bf16.h>
#include <math.h>

// Problem constants (from reference): BATCH=8, SEQ=2048, D_IN=D_OUT=1024,
// N_EXP=8, RANK=32, TOP_K=2
#define T_TOKENS 16384
#define D_IN     1024
#define D_OUT    1024
#define N_EXP    8
#define RANK     32
#define K_LORA   (N_EXP * RANK)   // 256
#define K_EXT    (D_IN + K_LORA)  // 1280

typedef __attribute__((ext_vector_type(16))) __bf16 v16bf;
typedef __attribute__((ext_vector_type(8)))  __bf16 v8bf;
typedef __attribute__((ext_vector_type(8)))  float  v8f;
typedef __attribute__((ext_vector_type(4)))  unsigned int u32x4;
typedef __attribute__((ext_vector_type(4)))  int i32x4;
typedef __attribute__((ext_vector_type(8)))  int i32x8;

// LDS staging geometry: tiles are 128 rows x 32 bf16, TDM-padded to an 80-byte
// row stride (64B data + 16B pad) => bank-conflict-free fragment reads.
#define ROW_US   40        // ushorts per padded LDS row (80 B)
#define TILE_US  (128 * ROW_US)          // 5120 ushorts = 10240 B per tile
#define STAGE_US (2 * TILE_US)           // A + B per stage
#define STAGE_B  (STAGE_US * 2)          // 20480 B
#define TILE_B   (TILE_US * 2)           // 10240 B

// ---------------- helpers ----------------

static __device__ __forceinline__ unsigned short f2bf(float f) {
  union { float f; unsigned u; } v; v.f = f;
  unsigned r = v.u + 0x7FFFu + ((v.u >> 16) & 1u);   // round-to-nearest-even
  return (unsigned short)(r >> 16);
}

static __device__ __forceinline__ float sigmoidf_(float x) {
  return 1.0f / (1.0f + __expf(-x));
}

// ---------------- TDM tile load ----------------
// 2D tensor_load_to_lds: tile_dim0 = 32 elems (64 B), tile_dim1 = 128 rows,
// data_size = 2 B, LDS padding: 16 B (pad_amount=3 -> 4 DWORDs) after every
// 64 B (pad_interval=3 -> 16 DWORDs).  Descriptor per CDNA5 ISA 8.3/8.4.
// This toolchain's builtin takes 6 args (g0, g1, g2, g3, extra, cpol).
static __device__ __forceinline__ void tdm_load_tile(unsigned lds_addr,
                                                     const unsigned short* gbase,
                                                     unsigned row_stride_elems) {
  unsigned long long ga = (unsigned long long)gbase;
  u32x4 g0;
  g0[0] = 1u;                                        // count=1, user mode
  g0[1] = lds_addr;                                  // LDS byte address
  g0[2] = (unsigned)ga;                              // global_addr[31:0]
  g0[3] = ((unsigned)(ga >> 32) & 0x01ffffffu)       // global_addr[56:32]
          | (2u << 30);                              // type = 2 ("image")
  i32x8 g1;
  g1[0] = (int)((1u << 16)                           // data_size = 2 B
                | (1u << 20)                         // pad_enable
                | (3u << 22)                         // pad_interval: 16 DWORDs
                | (3u << 25));                       // pad_amount: 4 DWORDs
  g1[1] = (int)((row_stride_elems & 0xffffu) << 16); // tensor_dim0 lo16
  g1[2] = (int)((row_stride_elems >> 16)             // tensor_dim0 hi16
                | (0x4000u << 16));                  // tensor_dim1 lo16 = 16384
  g1[3] = (int)(32u << 16);                          // tensor_dim1 hi | tile_dim0=32
  g1[4] = (int)128;                                  // tile_dim1=128, tile_dim2=0
  g1[5] = (int)row_stride_elems;                     // tensor_dim0_stride lo32
  g1[6] = 0;
  g1[7] = 0;
  i32x4 zz4 = {0, 0, 0, 0};
  i32x8 zz8 = {0, 0, 0, 0, 0, 0, 0, 0};
  __builtin_amdgcn_tensor_load_to_lds(g0, g1, zz4, zz4, zz8, 0);
}

// ---------------- LDS fragment loads (layouts per ISA 7.12.2) ----------------

// A-fragment (16x32 bf16, MxK): lane L -> row = L&15, h = L>>4,
// K chunks [8h, 8h+8) and [16+8h, 16+8h+8)  => two 16-B LDS reads.
static __device__ __forceinline__ v16bf lds_frag_a(const unsigned short* tile,
                                                   int row0, int lane) {
  int r = row0 + (lane & 15);
  int h = lane >> 4;
  const unsigned short* p = tile + r * ROW_US + 8 * h;
  v8bf lo = *(const v8bf*)(p);        // 16-B aligned: r*80 + 16h
  v8bf hi = *(const v8bf*)(p + 16);   // +32 B
  return __builtin_shufflevector(lo, hi, 0, 1, 2, 3, 4, 5, 6, 7,
                                 8, 9, 10, 11, 12, 13, 14, 15);
}

// B-fragment (32x16 bf16, KxN): lane L -> col = L&15, h = L>>4,
// K chunks [8h, 8h+8) and [16+8h, 16+8h+8)  => two 16-B LDS reads.
static __device__ __forceinline__ v16bf lds_frag_b(const unsigned short* tile,
                                                   int col0, int lane) {
  int c = col0 + (lane & 15);
  int h = lane >> 4;
  const unsigned short* p = tile + c * ROW_US + 8 * h;
  v8bf lo = *(const v8bf*)(p);
  v8bf hi = *(const v8bf*)(p + 16);
  return __builtin_shufflevector(lo, hi, 0, 1, 2, 3, 4, 5, 6, 7,
                                 8, 9, 10, 11, 12, 13, 14, 15);
}

// ---------------- shared TDM-fed GEMM main loop ----------------
// 8 waves (2x4): wave tile 64(M) x 32(N) = 4x2 WMMA accumulators.
// Wave 0 drives the TDM double-buffer; s_wait_tensorcnt + barrier per stage.
static __device__ __forceinline__ void gemm_tdm_loop(
    const unsigned short* Ag, int lda, int arow0,
    const unsigned short* Bg, int ldb, int bcol0,
    unsigned short* sh, int nk, v8f (&acc)[4][2]) {
  int lane = threadIdx.x & 31;
  int wave = threadIdx.x >> 5;
  int wm = (wave >> 2) * 64;
  int wn = (wave & 3) * 32;
  // LDS aperture keeps the LDS byte offset in addr[31:0] (ISA 10.2)
  unsigned lds0 = (unsigned)(unsigned long long)(void*)sh;

  if (wave == 0) {
    tdm_load_tile(lds0,          Ag + (size_t)arow0 * lda, (unsigned)lda);
    tdm_load_tile(lds0 + TILE_B, Bg + (size_t)bcol0 * ldb, (unsigned)ldb);
    __builtin_amdgcn_s_wait_tensorcnt(0);
  }
  __syncthreads();

  for (int ks = 0; ks < nk; ++ks) {
    int cur = ks & 1;
    if (wave == 0 && ks + 1 < nk) {   // DMA next stage while everyone computes
      unsigned nb = lds0 + (unsigned)((ks + 1) & 1) * STAGE_B;
      int k1 = (ks + 1) * 32;
      tdm_load_tile(nb,          Ag + (size_t)arow0 * lda + k1, (unsigned)lda);
      tdm_load_tile(nb + TILE_B, Bg + (size_t)bcol0 * ldb + k1, (unsigned)ldb);
    }

    const unsigned short* At = sh + cur * STAGE_US;
    const unsigned short* Bt = At + TILE_US;
    v16bf a[4], b[2];
#pragma unroll
    for (int im = 0; im < 4; ++im) a[im] = lds_frag_a(At, wm + im * 16, lane);
#pragma unroll
    for (int in = 0; in < 2; ++in) b[in] = lds_frag_b(Bt, wn + in * 16, lane);
#pragma unroll
    for (int im = 0; im < 4; ++im)
#pragma unroll
      for (int in = 0; in < 2; ++in)
        acc[im][in] = __builtin_amdgcn_wmma_f32_16x16x32_bf16(
            false, a[im], false, b[in], (short)0, acc[im][in], false, false);

    if (wave == 0 && ks + 1 < nk) __builtin_amdgcn_s_wait_tensorcnt(0);
    __syncthreads();
  }
}

// ---------------- conversion kernels ----------------

__global__ __launch_bounds__(256) void cvt_x_kernel(
    const float* __restrict__ x, unsigned short* __restrict__ aext, int total) {
  int idx = blockIdx.x * blockDim.x + threadIdx.x;
  if (idx >= total) return;
  int t = idx >> 10;        // D_IN == 1024
  int k = idx & 1023;
  aext[(size_t)t * K_EXT + k] = f2bf(x[idx]);
}

__global__ __launch_bounds__(256) void cvt_w_kernel(
    const float* __restrict__ baseW, const float* __restrict__ B,
    const float* __restrict__ Sb, unsigned short* __restrict__ wcat, int total) {
  int idx = blockIdx.x * blockDim.x + threadIdx.x;
  if (idx >= total) return;
  int o = idx / K_EXT;
  int c = idx - o * K_EXT;
  float v;
  if (c < D_IN) {
    v = baseW[(size_t)o * D_IN + c];
  } else {
    int j = c - D_IN;                                   // e*32 + r
    size_t bi = ((size_t)(j >> 5) * D_OUT + o) * RANK + (j & 31);
    v = B[bi] * sigmoidf_(Sb[bi]);
  }
  wcat[idx] = f2bf(v);
}

__global__ __launch_bounds__(256) void cvt_a_kernel(
    const float* __restrict__ A, const float* __restrict__ Sa,
    unsigned short* __restrict__ atw, int total) {
  int idx = blockIdx.x * blockDim.x + threadIdx.x;
  if (idx >= total) return;
  atw[idx] = f2bf(A[idx] * sigmoidf_(Sa[idx]));
}

// ---------------- router: logits -> softmax -> top-2 -> dense gate ----------------

__global__ __launch_bounds__(256) void router_kernel(
    const float* __restrict__ x, const float* __restrict__ rW,
    const float* __restrict__ rb, float* __restrict__ gate) {
  __shared__ float sW[N_EXP * D_IN];
  for (int i = threadIdx.x; i < N_EXP * D_IN; i += 256) sW[i] = rW[i];
  __syncthreads();

  int lane = threadIdx.x & 31;
  int wave = threadIdx.x >> 5;
  int t = blockIdx.x * 8 + wave;

  const float* xt = x + (size_t)t * D_IN;
  float acc[N_EXP];
#pragma unroll
  for (int e = 0; e < N_EXP; ++e) acc[e] = 0.0f;

  for (int k = lane; k < D_IN; k += 32) {
    float xv = xt[k];
#pragma unroll
    for (int e = 0; e < N_EXP; ++e)
      acc[e] = fmaf(xv, sW[e * D_IN + k], acc[e]);
  }
#pragma unroll
  for (int e = 0; e < N_EXP; ++e) {
#pragma unroll
    for (int off = 16; off > 0; off >>= 1)
      acc[e] += __shfl_xor(acc[e], off, 32);
  }

  if (lane == 0) {
    float l[N_EXP], p[N_EXP];
    float mx = -3.4e38f;
#pragma unroll
    for (int e = 0; e < N_EXP; ++e) { l[e] = acc[e] + rb[e]; mx = fmaxf(mx, l[e]); }
    float s = 0.0f;
#pragma unroll
    for (int e = 0; e < N_EXP; ++e) { p[e] = __expf(l[e] - mx); s += p[e]; }
    float inv = 1.0f / s;
#pragma unroll
    for (int e = 0; e < N_EXP; ++e) p[e] *= inv;
    int i0 = 0;
#pragma unroll
    for (int e = 1; e < N_EXP; ++e) if (p[e] > p[i0]) i0 = e;
    int i1 = (i0 == 0) ? 1 : 0;
#pragma unroll
    for (int e = 0; e < N_EXP; ++e)
      if (e != i0 && p[e] > p[i1]) i1 = e;
    float* g = gate + (size_t)t * N_EXP;
#pragma unroll
    for (int e = 0; e < N_EXP; ++e) g[e] = 0.0f;
    g[i0] = p[i0];
    g[i1] = p[i1];
  }
}

// ---------------- GEMM 1: h = x @ A_t^T, gated, bf16 into aext cols [1024,1280) ----

__global__ __launch_bounds__(256) void lora_h_gemm(
    unsigned short* aext, const unsigned short* __restrict__ atw,
    const float* __restrict__ gate) {
  __shared__ unsigned short sh[2 * STAGE_US];
  int lane = threadIdx.x & 31;
  int wave = threadIdx.x >> 5;

  v8f z = {0.f, 0.f, 0.f, 0.f, 0.f, 0.f, 0.f, 0.f};
  v8f acc[4][2];
#pragma unroll
  for (int im = 0; im < 4; ++im) { acc[im][0] = z; acc[im][1] = z; }

  gemm_tdm_loop(aext, K_EXT, blockIdx.x * 128,
                atw, D_IN, blockIdx.y * 128, sh, D_IN / 32, acc);

  // C layout: n = lane&15, m = 8*(lane>>4) + i (ISA 7.12.2)
  int bm = blockIdx.x * 128 + (wave >> 2) * 64;
  int bn = blockIdx.y * 128 + (wave & 3) * 32;
  int nloc = lane & 15;
  int mh = (lane >> 4) * 8;
#pragma unroll
  for (int im = 0; im < 4; ++im) {
#pragma unroll
    for (int in = 0; in < 2; ++in) {
      int col = bn + in * 16 + nloc;     // 0..255 -> expert = col>>5
      int e = col >> 5;
#pragma unroll
      for (int i = 0; i < 8; ++i) {
        int m = bm + im * 16 + mh + i;
        float g = gate[(size_t)m * N_EXP + e];
        aext[(size_t)m * K_EXT + D_IN + col] = f2bf(acc[im][in][i] * g);
      }
    }
  }
}

// ---------------- GEMM 2: y = [x | hg] @ [W ; B_t]^T + b (K = 1280) ----------------

__global__ __launch_bounds__(256) void out_gemm(
    const unsigned short* __restrict__ aext, const unsigned short* __restrict__ wcat,
    const float* __restrict__ bias, float* __restrict__ out) {
  __shared__ unsigned short sh[2 * STAGE_US];
  int lane = threadIdx.x & 31;
  int wave = threadIdx.x >> 5;

  v8f z = {0.f, 0.f, 0.f, 0.f, 0.f, 0.f, 0.f, 0.f};
  v8f acc[4][2];
#pragma unroll
  for (int im = 0; im < 4; ++im) { acc[im][0] = z; acc[im][1] = z; }

  gemm_tdm_loop(aext, K_EXT, blockIdx.x * 128,
                wcat, K_EXT, blockIdx.y * 128, sh, K_EXT / 32, acc);

  int bm = blockIdx.x * 128 + (wave >> 2) * 64;
  int bn = blockIdx.y * 128 + (wave & 3) * 32;
  int nloc = lane & 15;
  int mh = (lane >> 4) * 8;
#pragma unroll
  for (int im = 0; im < 4; ++im) {
#pragma unroll
    for (int in = 0; in < 2; ++in) {
      int col = bn + in * 16 + nloc;
      float bv = bias[col];
#pragma unroll
      for (int i = 0; i < 8; ++i) {
        int m = bm + im * 16 + mh + i;
        out[(size_t)m * D_OUT + col] = acc[im][in][i] + bv;
      }
    }
  }
}

// ---------------- host launcher ----------------

extern "C" void kernel_launch(void* const* d_in, const int* in_sizes, int n_in,
                              void* d_out, int out_size, void* d_ws, size_t ws_size,
                              hipStream_t stream) {
  (void)in_sizes; (void)n_in; (void)out_size; (void)ws_size;
  const float* x   = (const float*)d_in[0];
  const float* bW  = (const float*)d_in[1];
  const float* bb  = (const float*)d_in[2];
  const float* rW  = (const float*)d_in[3];
  const float* rb  = (const float*)d_in[4];
  const float* A   = (const float*)d_in[5];
  const float* Sa  = (const float*)d_in[6];
  const float* B   = (const float*)d_in[7];
  const float* Sb  = (const float*)d_in[8];
  float* out = (float*)d_out;

  // Workspace layout (bf16 staging; total ~45.6 MB)
  char* ws = (char*)d_ws;
  size_t off = 0;
  unsigned short* aext = (unsigned short*)(ws + off);  off += (size_t)T_TOKENS * K_EXT * 2;  // [T,1280]
  unsigned short* wcat = (unsigned short*)(ws + off);  off += (size_t)D_OUT * K_EXT * 2;     // [1024,1280]
  unsigned short* atw  = (unsigned short*)(ws + off);  off += (size_t)K_LORA * D_IN * 2;     // [256,1024]
  float* gate          = (float*)(ws + off);                                                 // [T,8]

  {
    int total = T_TOKENS * D_IN;
    cvt_x_kernel<<<(total + 255) / 256, 256, 0, stream>>>(x, aext, total);
  }
  {
    int total = D_OUT * K_EXT;
    cvt_w_kernel<<<(total + 255) / 256, 256, 0, stream>>>(bW, B, Sb, wcat, total);
  }
  {
    int total = K_LORA * D_IN;
    cvt_a_kernel<<<(total + 255) / 256, 256, 0, stream>>>(A, Sa, atw, total);
  }
  router_kernel<<<T_TOKENS / 8, 256, 0, stream>>>(x, rW, rb, gate);

  dim3 gh(T_TOKENS / 128, K_LORA / 128);   // (128, 2)
  lora_h_gemm<<<gh, 256, 0, stream>>>(aext, atw, gate);

  dim3 gy(T_TOKENS / 128, D_OUT / 128);    // (128, 8)
  out_gemm<<<gy, 256, 0, stream>>>(aext, wcat, bb, out);
}
